// Correlation_8555574854333
// MI455X (gfx1250) — matrix operand order repaired and checked by reading, not compiled, not run on hardware
//
#include <hip/hip_runtime.h>
#include <hip/hip_bf16.h>

// FlowNetC correlation / cost volume on gfx1250 (MI455X).
// out[b,h,w, (dy+4)*9+(dx+4)] = sum_c feat1[b,h,w,c] * feat2[b,h+dy,w+dx,c]
// Strategy: banded A*B^T per (b,h,dy) via v_wmma_f32_16x16x32_bf16, with f32
// emulated as bf16 hi/lo split (hi*hi + hi*lo + lo*hi), f32 accumulation.
// dy-loop is software-pipelined: feat2 tile for dy+1 is prefetched into
// registers while the WMMAs for dy execute (global-load latency hidden
// behind matrix math; LDS stays single-buffered).

#define BB 8
#define HH 128
#define WW 256
#define CC 128
#define MD 4
#define DD 9
#define OUTC 81
#define LSTR 136   // LDS row stride in bf16 (128 + 8 pad: 16B-aligned, bank-skewed)

typedef __attribute__((ext_vector_type(16))) __bf16 v16bf;
typedef __attribute__((ext_vector_type(8)))  __bf16 v8bf;
typedef __attribute__((ext_vector_type(8)))  float  v8f;
typedef __attribute__((ext_vector_type(4)))  float  v4f;

__device__ __forceinline__ v16bf cat8(v8bf a, v8bf b) {
  return __builtin_shufflevector(a, b, 0,1,2,3,4,5,6,7,8,9,10,11,12,13,14,15);
}

__launch_bounds__(128)
__global__ void corr_wmma_bf16x3_kernel(const float* __restrict__ f1,
                                        const float* __restrict__ f2,
                                        float* __restrict__ out) {
  // Block: 4 waves of 32. Tile: 32 pixels along W, all 81 displacements.
  __shared__ __align__(16) __bf16 s1h[32][LSTR];
  __shared__ __align__(16) __bf16 s1l[32][LSTR];
  __shared__ __align__(16) __bf16 s2h[48][LSTR];
  __shared__ __align__(16) __bf16 s2l[48][LSTR];

  const int tid  = threadIdx.x;
  const int lane = tid & 31;
  const int wave = tid >> 5;
  const int sub  = wave >> 1;   // which 16-pixel subtile (0/1)
  const int ct   = wave & 1;    // which 16-wide column tile of the 32-col band window
  const int hk   = lane >> 4;   // lane half (K-group select in fragments)
  const int ln   = lane & 15;

  const int w0 = blockIdx.x * 32;
  const int h  = blockIdx.y;
  const int b  = blockIdx.z;

  // ---- load feat1 tile: 32 pixels x 128 ch, split into bf16 hi/lo ----
  {
    const float* f1p = f1 + (((size_t)b * HH + h) * WW + w0) * CC;
#pragma unroll
    for (int i = 0; i < 8; ++i) {
      int q  = tid + i * 128;        // float4 index 0..1023
      int p  = q >> 5;               // pixel 0..31
      int c4 = (q & 31) << 2;        // channel
      v4f v = *(const v4f*)(f1p + (size_t)p * CC + c4);
#pragma unroll
      for (int k = 0; k < 4; ++k) {
        float  x  = v[k];
        __bf16 hi = (__bf16)x;
        s1h[p][c4 + k] = hi;
        s1l[p][c4 + k] = (__bf16)(x - (float)hi);
      }
    }
  }

  // A-fragment row and B-fragment column owned by this lane (fixed across dy)
  const int arow = 16 * sub + ln;           // feat1 pixel row in LDS
  const int bcol = 16 * sub + 16 * ct + ln; // feat2 column in 48-col LDS window
  float* const outp = out + ((((size_t)b * HH + h) * WW + w0 + 16 * sub) * OUTC);

  // Per-thread register staging for one feat2 window (48 cols x 128 ch / 128 thr)
  v4f pre[12];

  // Prefetch dy=0 window into registers.
  auto fetch = [&](int dyi) {
    const int hrow = h + dyi - MD;
    const bool hok = (unsigned)hrow < (unsigned)HH;
    const float* base = f2 + ((size_t)b * HH + hrow) * WW * CC;
#pragma unroll
    for (int i = 0; i < 12; ++i) {
      int q   = tid + i * 128;       // float4 index 0..1535
      int col = q >> 5;              // 0..47
      int c4  = (q & 31) << 2;
      int wp  = w0 - MD + col;
      v4f v = {0.f, 0.f, 0.f, 0.f};
      if (hok && (unsigned)wp < (unsigned)WW)
        v = *(const v4f*)(base + (size_t)wp * CC + c4);
      pre[i] = v;
    }
  };
  fetch(0);

  for (int dyi = 0; dyi < DD; ++dyi) {
    __syncthreads();  // previous iteration's s2 fully consumed

    // ---- convert staged registers -> bf16 hi/lo LDS window ----
#pragma unroll
    for (int i = 0; i < 12; ++i) {
      int q   = tid + i * 128;
      int col = q >> 5;
      int c4  = (q & 31) << 2;
      v4f v = pre[i];
#pragma unroll
      for (int k = 0; k < 4; ++k) {
        float  x  = v[k];
        __bf16 hi = (__bf16)x;
        s2h[col][c4 + k] = hi;
        s2l[col][c4 + k] = (__bf16)(x - (float)hi);
      }
    }
    __syncthreads();

    // ---- issue next window's global loads; they retire behind the WMMAs ----
    if (dyi + 1 < DD) fetch(dyi + 1);

    // ---- 16x16 output tile: D = A(16xC) * B(Cx16), C=128 in 4 K-steps of 32 ----
    v8f acc = {0.f, 0.f, 0.f, 0.f, 0.f, 0.f, 0.f, 0.f};
#pragma unroll
    for (int kb = 0; kb < 4; ++kb) {
      // A fragment (16-bit 16x32 layout): lane half selects K base {0,8}; runs at +0 and +16
      int ca = kb * 32 + (hk << 3);
      v16bf ah = cat8(*(const v8bf*)&s1h[arow][ca], *(const v8bf*)&s1h[arow][ca + 16]);
      v16bf al = cat8(*(const v8bf*)&s1l[arow][ca], *(const v8bf*)&s1l[arow][ca + 16]);
      // B fragment: lane half selects K base {0,8}; runs at +0 and +16 (same as A)
      int cb = kb * 32 + (hk << 3);
      v16bf bh = cat8(*(const v8bf*)&s2h[bcol][cb], *(const v8bf*)&s2h[bcol][cb + 16]);
      v16bf bl = cat8(*(const v8bf*)&s2l[bcol][cb], *(const v8bf*)&s2l[bcol][cb + 16]);

      acc = __builtin_amdgcn_wmma_f32_16x16x32_bf16(false, ah, false, bh, (short)0, acc, false, false);
      acc = __builtin_amdgcn_wmma_f32_16x16x32_bf16(false, ah, false, bl, (short)0, acc, false, false);
      acc = __builtin_amdgcn_wmma_f32_16x16x32_bf16(false, al, false, bh, (short)0, acc, false, false);
    }

    // ---- scatter the band diagonal: D[m,n] useful when dx4 = 16*ct + n - m in [0,8] ----
#pragma unroll
    for (int e = 0; e < 8; ++e) {
      int m   = hk * 8 + e;          // C/D layout: lanes 0-15 rows 0-7, lanes 16-31 rows 8-15
      int dx4 = 16 * ct + ln - m;
      if ((unsigned)dx4 <= 8u)
        outp[(size_t)m * OUTC + dyi * DD + dx4] = acc[e];
    }
  }
}

extern "C" void kernel_launch(void* const* d_in, const int* in_sizes, int n_in,
                              void* d_out, int out_size, void* d_ws, size_t ws_size,
                              hipStream_t stream) {
  const float* f1 = (const float*)d_in[0];
  const float* f2 = (const float*)d_in[1];
  float* out = (float*)d_out;
  dim3 grid(WW / 32, HH, BB);  // (8, 128, 8) blocks
  dim3 block(128);
  hipLaunchKernelGGL(corr_wmma_bf16x3_kernel, grid, block, 0, stream, f1, f2, out);
}
